// TransformerEncoder_5428838662355
// MI455X (gfx1250) — compile-verified, matrix operand order checked by reading, not run on hardware
//
#include <hip/hip_runtime.h>
#include <math.h>

// ---------- types ----------
typedef __attribute__((ext_vector_type(16))) __bf16 v16bf;
typedef __attribute__((ext_vector_type(8)))  float  v8f;
typedef __attribute__((ext_vector_type(4)))  unsigned int v4u;
typedef __attribute__((ext_vector_type(8)))  unsigned int v8u;
typedef unsigned short ush;
typedef unsigned int   u32;
typedef unsigned long long u64;

#define TOKENS 8192
#define DMODEL 768
#define NHEAD  12
#define DHEAD  64
#define DMLP   3072
#define SEQ    1024

// fp32 -> bf16 round-to-nearest-even (bit trick)
static __device__ __forceinline__ ush f2bf(float f) {
    union { float f; u32 u; } c; c.f = f;
    u32 r = c.u + 0x7FFFu + ((c.u >> 16) & 1u);
    return (ush)(r >> 16);
}

// ---------- CDNA5 sync / async helpers ----------
static __device__ __forceinline__ void wait_tensor0() {
#if __has_builtin(__builtin_amdgcn_s_wait_tensorcnt)
    __builtin_amdgcn_s_wait_tensorcnt(0);
#else
    asm volatile("s_wait_tensorcnt 0x0" ::: "memory");
#endif
}
static __device__ __forceinline__ void wait_async0() {
    asm volatile("s_wait_asynccnt 0x0" ::: "memory");
}
// Per-lane async copy: 16B global -> LDS (ASYNCcnt-tracked)
static __device__ __forceinline__ void async_ld_b128(const ush* gptr, const ush* lptr) {
    u32 lds = (u32)(uintptr_t)lptr;
    u64 ga  = (u64)(uintptr_t)gptr;
    asm volatile("global_load_async_to_lds_b128 %0, %1, off"
                 :: "v"(lds), "v"(ga) : "memory");
}
// TDM: load a 2-D bf16 tile (tile_w x tile_h elements) from a row-major
// tensor (ld = row stride in elements) into LDS, padding LDS rows per the
// D# group1 pad codes so rows land at a bank-friendly stride.
static __device__ __forceinline__ void tdm_load_tile_2d(
    const ush* gsrc, u32 lds_addr, u32 ld, u32 rows_total,
    u32 tile_w, u32 tile_h, u32 pad_int_code, u32 pad_amt_code) {
    u64 ga = (u64)(uintptr_t)gsrc;
    v4u g0; v8u g1;
    g0[0] = 1u;                                   // count=1 (valid user D#)
    g0[1] = lds_addr;                             // lds_addr (bytes)
    g0[2] = (u32)ga;                              // global_addr[95:64]
    g0[3] = ((u32)(ga >> 32) & 0x01FFFFFFu) | (2u << 30);  // addr msbs | type=2
    g1[0] = (1u << 16)                            // data_size = 1 -> 2 bytes
          | (1u << 20)                            // pad_enable
          | (pad_int_code << 22)                  // pad interval code
          | (pad_amt_code << 25);                 // pad amount code
    g1[1] = (ld & 0xFFFFu) << 16;                 // tensor_dim0 lo16 (abar=0)
    g1[2] = (ld >> 16) | ((rows_total & 0xFFFFu) << 16);   // dim0 hi | dim1 lo
    g1[3] = (rows_total >> 16) | (tile_w << 16);  // dim1 hi | tile_dim0
    g1[4] = tile_h;                               // tile_dim1 (tile_dim2=0)
    g1[5] = ld;                                   // tensor_dim0_stride lo32
    g1[6] = 0u;
    g1[7] = 0u;
    asm volatile("tensor_load_to_lds %0, %1" :: "s"(g0), "s"(g1) : "memory");
}

// ---------- WMMA fragment ----------
// CDNA5 16-bit A-matrix 16x32 layout (ISA 7.12.2):
//  lanes 0-15 : row M=lane,    K = {0..7, 16..23}
//  lanes 16-31: row M=lane-16, K = {8..15, 24..31}
// B tiles are stored TRANSPOSED ([N][K]) so the gather is identical.
struct Frag {
    union { v16bf v; u32 u[8]; ush s[16]; };
};

static __device__ __forceinline__ Frag frag_load(const ush* base, int stride) {
    const int lane = threadIdx.x & 31;
    const int g    = (lane >> 4) & 1;
    const ush* p   = base + (size_t)(lane & 15) * stride + g * 8;
    Frag f;
#pragma unroll
    for (int i = 0; i < 4; ++i) {
        f.u[i]     = *(const u32*)(p + 2 * i);
        f.u[4 + i] = *(const u32*)(p + 16 + 2 * i);
    }
    return f;
}

static __device__ __forceinline__ v8f wmma_bf16(const Frag& a, const Frag& b, v8f c) {
    return __builtin_amdgcn_wmma_f32_16x16x32_bf16(false, a.v, false, b.v,
                                                   (short)0, c, false, false);
}

static __device__ __forceinline__ float gelu_exact(float x) {
    return 0.5f * x * (1.0f + erff(x * 0.70710678118654752f));
}

// ---------- fp32 -> bf16 bulk convert ----------
__global__ void cvt_f32_bf16(const float* __restrict__ src, ush* __restrict__ dst, int n) {
    for (int i = blockIdx.x * blockDim.x + threadIdx.x; i < n; i += gridDim.x * blockDim.x)
        dst[i] = f2bf(src[i]);
}

// ---------- LayerNorm (one row of D=768 per block) -> bf16 ----------
__global__ __launch_bounds__(256)
void layernorm_bf16(const float* __restrict__ x, const float* __restrict__ g,
                    const float* __restrict__ b, ush* __restrict__ out) {
    __shared__ float r1[256], r2[256];
    const int tid = threadIdx.x;
    const float* xr = x + (size_t)blockIdx.x * DMODEL;
    float s = 0.f, s2 = 0.f;
#pragma unroll
    for (int i = tid; i < DMODEL; i += 256) { float v = xr[i]; s += v; s2 += v * v; }
    r1[tid] = s; r2[tid] = s2; __syncthreads();
    for (int st = 128; st > 0; st >>= 1) {
        if (tid < st) { r1[tid] += r1[tid + st]; r2[tid] += r2[tid + st]; }
        __syncthreads();
    }
    const float mu  = r1[0] * (1.0f / DMODEL);
    const float var = r2[0] * (1.0f / DMODEL) - mu * mu;
    const float rs  = rsqrtf(var + 1e-5f);
    ush* orow = out + (size_t)blockIdx.x * DMODEL;
#pragma unroll
    for (int i = tid; i < DMODEL; i += 256)
        orow[i] = f2bf((xr[i] - mu) * rs * g[i] + b[i]);
}

// ---------- WMMA GEMM: C[M,N] = A[M,K](bf16) * B[K,N](bf16) + bias (+res)(+gelu) ----------
// BM=BN=64, BK=64, 128 threads (4 waves), each wave a 32x32 region.
// Double-buffered pipeline: TDM stages the next A tile and global loads stage
// the next B tile while the current tile's 8 WMMAs execute.
template <bool GELU, bool RES, bool F32OUT>
__global__ __launch_bounds__(128)
void gemm_bf16(const ush* __restrict__ A, const ush* __restrict__ B,
               const float* __restrict__ bias, const float* __restrict__ res,
               float* __restrict__ outF, ush* __restrict__ outH,
               int M, int N, int K) {
    __shared__ ush As[2][64][72];   // [m][k]  (TDM pad: 32 DW rows + 4 DW pad)
    __shared__ ush Bs[2][64][72];   // transposed: [n][k]
    const int tid  = threadIdx.x;
    const int wave = tid >> 5;
    const int lane = tid & 31;
    const int m0 = blockIdx.y * 64, n0 = blockIdx.x * 64;
    const int wr = (wave >> 1) * 32, wc = (wave & 1) * 32;
    const int nk = K >> 6;

    v8f acc[2][2];
#pragma unroll
    for (int i = 0; i < 2; ++i)
#pragma unroll
        for (int j = 0; j < 2; ++j) acc[i][j] = {};

    // ---- prologue: stage tile 0 ----
    if (wave == 0)
        tdm_load_tile_2d(A + (size_t)m0 * K, (u32)(uintptr_t)&As[0][0][0],
                         (u32)K, (u32)M, 64u, 64u, 4u, 3u);
    uint4 breg[4];
#pragma unroll
    for (int i = 0; i < 4; ++i) {
        int c = tid + i * 128;
        int kk = c >> 3, n8 = (c & 7) * 8;
        breg[i] = *(const uint4*)&B[(size_t)kk * N + n0 + n8];
    }

    for (int it = 0; it < nk; ++it) {
        const int cur = it & 1;
        const int k0  = it << 6;
        // scatter staged B registers into Bs[cur] (transpose)
#pragma unroll
        for (int i = 0; i < 4; ++i) {
            int c = tid + i * 128;
            int kk = c >> 3, n8 = (c & 7) * 8;
            union { uint4 q; ush s[8]; } t; t.q = breg[i];
#pragma unroll
            for (int j = 0; j < 8; ++j) Bs[cur][n8 + j][kk] = t.s[j];
        }
        if (wave == 0) wait_tensor0();   // A[cur] is the only outstanding TDM
        __syncthreads();

        // issue next tile's copies; they overlap the WMMAs below
        if (it + 1 < nk) {
            const int k1 = k0 + 64;
            if (wave == 0)
                tdm_load_tile_2d(A + (size_t)m0 * K + k1,
                                 (u32)(uintptr_t)&As[cur ^ 1][0][0],
                                 (u32)K, (u32)M, 64u, 64u, 4u, 3u);
#pragma unroll
            for (int i = 0; i < 4; ++i) {
                int c = tid + i * 128;
                int kk = c >> 3, n8 = (c & 7) * 8;
                breg[i] = *(const uint4*)&B[(size_t)(k1 + kk) * N + n0 + n8];
            }
            if (k1 + 64 < K) {   // warm L2 two slabs ahead for the TDM
                __builtin_prefetch(&A[(size_t)(m0 + (tid >> 1)) * K + k1 + 64], 0, 1);
                __builtin_prefetch(&B[(size_t)(k1 + 64 + (tid >> 1)) * N + n0], 0, 1);
            }
        }

#pragma unroll
        for (int kk = 0; kk < 64; kk += 32) {
            Frag a0 = frag_load(&As[cur][wr][kk],      72);
            Frag a1 = frag_load(&As[cur][wr + 16][kk], 72);
            Frag b0 = frag_load(&Bs[cur][wc][kk],      72);
            Frag b1 = frag_load(&Bs[cur][wc + 16][kk], 72);
            acc[0][0] = wmma_bf16(a0, b0, acc[0][0]);
            acc[0][1] = wmma_bf16(a0, b1, acc[0][1]);
            acc[1][0] = wmma_bf16(a1, b0, acc[1][0]);
            acc[1][1] = wmma_bf16(a1, b1, acc[1][1]);
        }
        __syncthreads();
    }

    // epilogue: C layout — lane(0-15): col=lane, row=r; lane(16-31): col=lane-16, row=r+8
    const int lcol = lane & 15, lhi = (lane >> 4) * 8;
#pragma unroll
    for (int i = 0; i < 2; ++i) {
#pragma unroll
        for (int j = 0; j < 2; ++j) {
            const int n = n0 + wc + j * 16 + lcol;
            const float bv = bias[n];
#pragma unroll
            for (int r = 0; r < 8; ++r) {
                const int m = m0 + wr + i * 16 + lhi + r;
                float vv = acc[i][j][r] + bv;
                if (GELU) vv = gelu_exact(vv);
                if (RES)  vv += res[(size_t)m * N + n];
                if (F32OUT) outF[(size_t)m * N + n] = vv;
                else        outH[(size_t)m * N + n] = f2bf(vv);
            }
        }
    }
}

// ---------- Flash attention (bf16 in/out), scale = 1/sqrt(768) ----------
// Block: 64 queries of one (batch, head); 4 waves, 16 queries each.
// S^T = K*Q^T (per-lane column softmax stats), O^T = V^T * P^T; the exp'd S^T
// C-tiles remap ELEMENTWISE into B-fragments (no cross-lane transpose).
// Double-buffered: async K copies (ASYNCcnt) + register-staged V transpose
// for tile t+1 overlap tile t's WMMAs.
__global__ __launch_bounds__(128)
void attention_wmma(const ush* __restrict__ q, const ush* __restrict__ k,
                    const ush* __restrict__ v, ush* __restrict__ o) {
    __shared__ ush Ks[2][32][72];   // [key][feature]
    __shared__ ush Vs[2][64][40];   // transposed: [feature][key]
    const int qb = blockIdx.x, h = blockIdx.y, bb = blockIdx.z;
    const int tid = threadIdx.x, wave = tid >> 5, lane = tid & 31, l = lane & 15;
    const size_t rowbase = (size_t)bb * SEQ;
    const int col0  = h * DHEAD;
    const int qrow0 = qb * 64 + wave * 16;
    const float scale = 0.03608439182435161f;  // 1/sqrt(768)

    const ush* qp = q + (rowbase + qrow0) * DMODEL + col0;
    Frag bq0 = frag_load(qp, DMODEL);
    Frag bq1 = frag_load(qp + 32, DMODEL);

    v8f oacc[4];
#pragma unroll
    for (int t = 0; t < 4; ++t) oacc[t] = {};
    float mrow = -INFINITY, lsum = 0.f;

    // ---- prologue: stage key tile 0 ----
    uint4 vreg[2];
#pragma unroll
    for (int i = 0; i < 2; ++i) {
        int c = tid + i * 128;
        int kr = c >> 3, f8 = (c & 7) * 8;
        async_ld_b128(&k[(rowbase + kr) * DMODEL + col0 + f8], &Ks[0][kr][f8]);
        vreg[i] = *(const uint4*)&v[(rowbase + kr) * DMODEL + col0 + f8];
    }

    const int nkt = SEQ / 32;
    for (int it = 0; it < nkt; ++it) {
        const int cur = it & 1;
        const int kt  = it * 32;
        // scatter staged V registers into Vs[cur] (transpose)
#pragma unroll
        for (int i = 0; i < 2; ++i) {
            int c = tid + i * 128;
            int kr = c >> 3, f8 = (c & 7) * 8;
            union { uint4 q4; ush s[8]; } t; t.q4 = vreg[i];
#pragma unroll
            for (int j = 0; j < 8; ++j) Vs[cur][f8 + j][kr] = t.s[j];
        }
        wait_async0();     // K[cur] async copies are the only outstanding ones
        __syncthreads();

        // issue next key tile's copies; they overlap the WMMAs below
        if (it + 1 < nkt) {
            const int kn = kt + 32;
#pragma unroll
            for (int i = 0; i < 2; ++i) {
                int c = tid + i * 128;
                int kr = c >> 3, f8 = (c & 7) * 8;
                async_ld_b128(&k[(rowbase + kn + kr) * DMODEL + col0 + f8],
                              &Ks[cur ^ 1][kr][f8]);
                vreg[i] = *(const uint4*)&v[(rowbase + kn + kr) * DMODEL + col0 + f8];
            }
        }

        Frag aK00 = frag_load(&Ks[cur][0][0],  72);
        Frag aK01 = frag_load(&Ks[cur][0][32], 72);
        Frag aK10 = frag_load(&Ks[cur][16][0], 72);
        Frag aK11 = frag_load(&Ks[cur][16][32], 72);
        v8f s0 = {}; s0 = wmma_bf16(aK00, bq0, s0); s0 = wmma_bf16(aK01, bq1, s0);
        v8f s1 = {}; s1 = wmma_bf16(aK10, bq0, s1); s1 = wmma_bf16(aK11, bq1, s1);

        float sv[16];
#pragma unroll
        for (int r = 0; r < 8; ++r) { sv[r] = s0[r] * scale; sv[8 + r] = s1[r] * scale; }
        float tmax = sv[0];
#pragma unroll
        for (int i = 1; i < 16; ++i) tmax = fmaxf(tmax, sv[i]);
        tmax = fmaxf(tmax, __shfl_xor(tmax, 16));
        const float mnew  = fmaxf(mrow, tmax);
        const float alpha = __expf(mrow - mnew);
        float psum = 0.f;
#pragma unroll
        for (int i = 0; i < 16; ++i) { sv[i] = __expf(sv[i] - mnew); psum += sv[i]; }
        psum += __shfl_xor(psum, 16);
        lsum = lsum * alpha + psum;
        mrow = mnew;
#pragma unroll
        for (int t = 0; t < 4; ++t)
#pragma unroll
            for (int r = 0; r < 8; ++r) oacc[t][r] *= alpha;

        Frag pf;
#pragma unroll
        for (int i = 0; i < 16; ++i) pf.s[i] = f2bf(sv[i]);

#pragma unroll
        for (int t = 0; t < 4; ++t) {
            Frag av = frag_load(&Vs[cur][t * 16][0], 40);
            oacc[t] = wmma_bf16(av, pf, oacc[t]);
        }
        __syncthreads();
    }

    const float inv = 1.f / lsum;
    const size_t orow = (rowbase + qrow0 + l) * DMODEL + col0;
    const int fhi = (lane >> 4) * 8;
#pragma unroll
    for (int t = 0; t < 4; ++t) {
        union { uint4 q4; ush s[8]; } t8;
#pragma unroll
        for (int r = 0; r < 8; ++r) t8.s[r] = f2bf(oacc[t][r] * inv);
        *(uint4*)&o[orow + t * 16 + fhi] = t8.q4;
    }
}

// ---------- host launch ----------
extern "C" void kernel_launch(void* const* d_in, const int* in_sizes, int n_in,
                              void* d_out, int out_size, void* d_ws, size_t ws_size,
                              hipStream_t stream) {
    const float* x     = (const float*)d_in[0];
    const float* ln1_g = (const float*)d_in[1];
    const float* ln1_b = (const float*)d_in[2];
    const float* Wq    = (const float*)d_in[3];
    const float* bq    = (const float*)d_in[4];
    const float* Wk    = (const float*)d_in[5];
    const float* bk    = (const float*)d_in[6];
    const float* Wv    = (const float*)d_in[7];
    const float* bv    = (const float*)d_in[8];
    const float* Wo    = (const float*)d_in[9];
    const float* bo    = (const float*)d_in[10];
    const float* ln2_g = (const float*)d_in[11];
    const float* ln2_b = (const float*)d_in[12];
    const float* W1    = (const float*)d_in[13];
    const float* b1    = (const float*)d_in[14];
    const float* W2    = (const float*)d_in[15];
    const float* b2    = (const float*)d_in[16];

    char* w = (char*)d_ws;
    auto carve = [&](size_t bytes) -> void* {
        void* p = (void*)w;
        w += (bytes + 255) & ~(size_t)255;
        return p;
    };
    const size_t DD = (size_t)DMODEL * DMODEL;
    const size_t DM = (size_t)DMODEL * DMLP;
    const size_t TD = (size_t)TOKENS * DMODEL;
    const size_t TM = (size_t)TOKENS * DMLP;

    ush* wWq = (ush*)carve(DD * 2);
    ush* wWk = (ush*)carve(DD * 2);
    ush* wWv = (ush*)carve(DD * 2);
    ush* wWo = (ush*)carve(DD * 2);
    ush* wW1 = (ush*)carve(DM * 2);
    ush* wW2 = (ush*)carve(DM * 2);
    ush* h1  = (ush*)carve(TD * 2);
    ush* qb_ = (ush*)carve(TD * 2);
    ush* kb_ = (ush*)carve(TD * 2);
    ush* vb_ = (ush*)carve(TD * 2);
    ush* ab_ = (ush*)carve(TD * 2);
    float* out1 = (float*)carve(TD * 4);
    ush* h2  = (ush*)carve(TD * 2);
    ush* mid = (ush*)carve(TM * 2);

    cvt_f32_bf16<<<1024, 256, 0, stream>>>(Wq, wWq, (int)DD);
    cvt_f32_bf16<<<1024, 256, 0, stream>>>(Wk, wWk, (int)DD);
    cvt_f32_bf16<<<1024, 256, 0, stream>>>(Wv, wWv, (int)DD);
    cvt_f32_bf16<<<1024, 256, 0, stream>>>(Wo, wWo, (int)DD);
    cvt_f32_bf16<<<1024, 256, 0, stream>>>(W1, wW1, (int)DM);
    cvt_f32_bf16<<<1024, 256, 0, stream>>>(W2, wW2, (int)DM);

    layernorm_bf16<<<TOKENS, 256, 0, stream>>>(x, ln1_g, ln1_b, h1);

    dim3 gD(DMODEL / 64, TOKENS / 64);
    gemm_bf16<false, false, false><<<gD, 128, 0, stream>>>(
        h1, wWq, bq, nullptr, nullptr, qb_, TOKENS, DMODEL, DMODEL);
    gemm_bf16<false, false, false><<<gD, 128, 0, stream>>>(
        h1, wWk, bk, nullptr, nullptr, kb_, TOKENS, DMODEL, DMODEL);
    gemm_bf16<false, false, false><<<gD, 128, 0, stream>>>(
        h1, wWv, bv, nullptr, nullptr, vb_, TOKENS, DMODEL, DMODEL);

    attention_wmma<<<dim3(SEQ / 64, NHEAD, 8), 128, 0, stream>>>(qb_, kb_, vb_, ab_);

    gemm_bf16<false, true, true><<<gD, 128, 0, stream>>>(
        ab_, wWo, bo, x, out1, nullptr, TOKENS, DMODEL, DMODEL);

    layernorm_bf16<<<TOKENS, 256, 0, stream>>>(out1, ln2_g, ln2_b, h2);

    dim3 gM(DMLP / 64, TOKENS / 64);
    gemm_bf16<true, false, false><<<gM, 128, 0, stream>>>(
        h2, wW1, b1, nullptr, nullptr, mid, TOKENS, DMLP, DMODEL);

    gemm_bf16<true, true, true><<<gD, 128, 0, stream>>>(
        mid, wW2, b2, out1, (float*)d_out, nullptr, TOKENS, DMODEL, DMLP);
}
